// ReasoningCell_54331336295036
// MI455X (gfx1250) — compile-verified, matrix operand order checked by reading, not run on hardware
//
#include <hip/hip_runtime.h>
#include <hip/hip_bf16.h>
#include <math.h>

// ---------------------------------------------------------------------------
// MI455X (gfx1250) implementation of the MAC-style reasoning cell.
//
// Roofline: after factoring K/V projections out of the single-query attentions
// (score = x . (Wk_h^T qh), out_h = Wv_h (sum_j a_j x_j) + bv_h), the
// workload is memory-bound: ~1.9 GB of feature reads (~80 us @ 23.3 TB/s) and
// ~30 GFLOP of GEMM. All GEMMs run on v_wmma_f32_16x16x32_bf16 (f32 accum).
// B tiles are staged through LDS in WMMA fragment layout (bf16), shared by
// all 4 waves of a block; A fragments convert f32->bf16 in registers.
// ---------------------------------------------------------------------------

constexpr int CB  = 256;   // batch
constexpr int CD  = 1024;  // model dim
constexpr int CS  = 128;   // question length
constexpr int CNO = 256;   // objects
constexpr int CNR = 512;   // relations
constexpr int CNH = 16;    // heads (hd = 64)

typedef __attribute__((ext_vector_type(16))) __bf16 v16bf;
typedef __attribute__((ext_vector_type(8)))  float  v8f;

__device__ __forceinline__ float act_gelu(float x) {
  return 0.5f * x * (1.0f + erff(x * 0.7071067811865475f));
}
__device__ __forceinline__ float act_sigmoid(float x) {
  return 1.0f / (1.0f + expf(-x));
}

// ---------------------------------------------------------------------------
// Batched-strided GEMM:  O = act(alpha * A @ op(W) + bias)
//   A: [M x K] f32 (row stride lda, batch stride sA)
//   NTF=true  -> W is [N x K] row-major (x @ W.T)
//   NTF=false -> W is [K x N] row-major (x @ W)
// Block = 128 threads = 4 waves; block tile 64(M) x 16*TMAX(N).
// B tile (32 x 16*TMAX) is cooperatively converted to bf16 and stored in LDS
// pre-swizzled into the ISA's 16-bit B-fragment layout, double-buffered.
// Staging gathers all float4s before converting so global loads pipeline.
// ---------------------------------------------------------------------------
template<int TMAX, bool NTF, int ACT>
__global__ __launch_bounds__(128) void gemm_wmma_bf16_t(
    const float* __restrict__ A, long long lda, long long sA,
    const float* __restrict__ W, long long ldw, long long sW,
    const float* __restrict__ bias, long long sBias,
    float* __restrict__ O, long long ldo, long long sO,
    int M, int N, int K, float alpha)
{
  constexpr int NT16 = TMAX * 16;               // tile width (n)
  constexpr int TILE_ELEMS = 32 * NT16;         // f32 elements per B tile
  constexpr int STAGE_ITERS = TILE_ELEMS / (128 * 4);  // 4 (TMAX=4) or 1
  __shared__ __align__(32) __bf16 bsh[2][TMAX * 32 * 16];

  const int tid  = threadIdx.x;
  const int lane = tid & 31;
  const int wave = tid >> 5;
  const int z    = blockIdx.z;
  A += (long long)z * sA;
  W += (long long)z * sW;
  O += (long long)z * sO;
  const float* bptr = bias ? (bias + (long long)z * sBias) : nullptr;

  const int m0 = blockIdx.y * 64 + wave * 16;
  const bool active = (m0 < M);                 // wave-uniform
  const int n0 = blockIdx.x * NT16;
  const int ln = lane & 15;
  const int hf = lane >> 4;

  // Precompute staging coordinates (loop-invariant).
  int sg_n[STAGE_ITERS], sg_k[STAGE_ITERS], sg_lofs[STAGE_ITERS];
#pragma unroll
  for (int it = 0; it < STAGE_ITERS; ++it) {
    const int i = tid * 4 + it * 512;
    int n, k;
    if constexpr (NTF) { n = i >> 5; k = i & 31; }
    else               { k = i / NT16; n = i - (i / NT16) * NT16; }
    sg_n[it] = n; sg_k[it] = k;
    const int t   = n >> 4;
    const int col = n & 15;
    const int hfk = (k >> 3) & 1;
    const int e   = (k & 7) + ((k >> 4) << 3);
    sg_lofs[it] = ((t * 32) + (hfk << 4) + col) * 16 + e;
  }

  v8f acc[TMAX];
#pragma unroll
  for (int t = 0; t < TMAX; ++t) acc[t] = (v8f){0.f,0.f,0.f,0.f,0.f,0.f,0.f,0.f};

  const float* arow = A + (long long)(active ? (m0 + ln) : 0) * lda;

  int buf = 0;
  for (int kk = 0; kk < K; kk += 32, buf ^= 1) {
    // ---- stage B tile into LDS in fragment layout (bf16) ----
    float4 vv[STAGE_ITERS];
#pragma unroll
    for (int it = 0; it < STAGE_ITERS; ++it) {
      if constexpr (NTF) {
        // W[n][k]: 4 consecutive k of one row -> coalesced float4
        vv[it] = *reinterpret_cast<const float4*>(
            W + (long long)(n0 + sg_n[it]) * ldw + kk + sg_k[it]);
      } else {
        // W[k][n]: 4 consecutive n of one k-row -> coalesced float4
        vv[it] = *reinterpret_cast<const float4*>(
            W + (long long)(kk + sg_k[it]) * ldw + n0 + sg_n[it]);
      }
    }
#pragma unroll
    for (int it = 0; it < STAGE_ITERS; ++it) {
      __bf16* dst = &bsh[buf][sg_lofs[it]];
      if constexpr (NTF) {
        // 4 consecutive e slots -> contiguous 8B store
        dst[0] = (__bf16)vv[it].x; dst[1] = (__bf16)vv[it].y;
        dst[2] = (__bf16)vv[it].z; dst[3] = (__bf16)vv[it].w;
      } else {
        // 4 consecutive n -> 4 lane slots, stride 16 elements
        dst[0]  = (__bf16)vv[it].x; dst[16] = (__bf16)vv[it].y;
        dst[32] = (__bf16)vv[it].z; dst[48] = (__bf16)vv[it].w;
      }
    }
    __syncthreads();   // one barrier per K-step (double-buffered LDS)

    if (active) {
      if (kk + 32 < K) __builtin_prefetch(arow + kk + 32, 0, 1);
      // A fragment first (global latency overlaps the LDS reads below).
      const float4* p = reinterpret_cast<const float4*>(arow + kk + 8 * hf);
      const float4 x0 = p[0], x1 = p[1];   // k + 0..7
      const float4 y0 = p[4], y1 = p[5];   // k + 16..23

      // Preload ALL B fragments -> ds_load_b128s in flight together.
      v16bf bfr[TMAX];
#pragma unroll
      for (int t = 0; t < TMAX; ++t)
        bfr[t] = *reinterpret_cast<const v16bf*>(&bsh[buf][(t * 32 + lane) * 16]);

      v16bf a;
      a[0]=(__bf16)x0.x;  a[1]=(__bf16)x0.y;  a[2]=(__bf16)x0.z;  a[3]=(__bf16)x0.w;
      a[4]=(__bf16)x1.x;  a[5]=(__bf16)x1.y;  a[6]=(__bf16)x1.z;  a[7]=(__bf16)x1.w;
      a[8]=(__bf16)y0.x;  a[9]=(__bf16)y0.y;  a[10]=(__bf16)y0.z; a[11]=(__bf16)y0.w;
      a[12]=(__bf16)y1.x; a[13]=(__bf16)y1.y; a[14]=(__bf16)y1.z; a[15]=(__bf16)y1.w;

#pragma unroll
      for (int t = 0; t < TMAX; ++t)
        acc[t] = __builtin_amdgcn_wmma_f32_16x16x32_bf16(
            false, a, false, bfr[t], (short)0, acc[t], false, false);
    }
  }
  if (!active) return;   // no barriers below

  // Epilogue: element r of lane -> row m0 + r + 8*hf, col n0 + 16t + ln
#pragma unroll
  for (int t = 0; t < TMAX; ++t) {
    const int col = n0 + t * 16 + ln;
    const float bv = bptr ? bptr[col] : 0.f;
#pragma unroll
    for (int r = 0; r < 8; ++r) {
      const int row = m0 + r + 8 * hf;
      float v = acc[t][r] * alpha + bv;
      if constexpr (ACT == 1) v = act_gelu(v);
      if constexpr (ACT == 2) v = act_sigmoid(v);
      O[(long long)row * ldo + col] = v;
    }
  }
}

// ---------------------------------------------------------------------------
// Softmax over keys for 16 heads. scores [B, S, 16] -> attn [B, 16, S].
// One wave per (b, h); wave32 shuffle reductions.
// ---------------------------------------------------------------------------
__global__ __launch_bounds__(128) void softmax16_k(
    const float* __restrict__ sc, const int* __restrict__ mask,
    float* __restrict__ attn, int S)
{
  const int lane = threadIdx.x & 31;
  const int h = blockIdx.x * 4 + (threadIdx.x >> 5);
  const int b = blockIdx.y;
  const float* s = sc + (long long)b * S * CNH + h;

  float vals[16];   // S <= 512 -> at most 16 per lane
  int cnt = 0;
  float mx = -1e30f;
  for (int j = lane; j < S; j += 32) {
    float v = s[(long long)j * CNH];
    if (mask && mask[(long long)b * S + j] == 0) v = -1e9f;
    vals[cnt++] = v;
    mx = fmaxf(mx, v);
  }
  for (int o = 16; o; o >>= 1) mx = fmaxf(mx, __shfl_xor(mx, o, 32));
  float sum = 0.f;
  for (int i = 0; i < cnt; ++i) { vals[i] = expf(vals[i] - mx); sum += vals[i]; }
  for (int o = 16; o; o >>= 1) sum += __shfl_xor(sum, o, 32);
  const float inv = 1.0f / sum;
  float* ap = attn + ((long long)b * CNH + h) * S;
  cnt = 0;
  for (int j = lane; j < S; j += 32) ap[j] = vals[cnt++] * inv;
}

// Head-averaged attention weights (torch MHA returns these): out[b, j].
__global__ void attn_mean16_k(const float* __restrict__ attn,
                              float* __restrict__ out, int S)
{
  const int idx = blockIdx.x * blockDim.x + threadIdx.x;
  if (idx >= CB * S) return;
  const int b = idx / S, j = idx - b * S;
  const float* ap = attn + (long long)b * CNH * S + j;
  float s = 0.f;
#pragma unroll
  for (int h = 0; h < CNH; ++h) s += ap[(long long)h * S];
  out[idx] = s * (1.0f / CNH);
}

__global__ void concat2_k(const float* __restrict__ a, const float* __restrict__ b,
                          float* __restrict__ o, int rows, int d)
{
  const long long i = (long long)blockIdx.x * blockDim.x + threadIdx.x;
  if (i >= (long long)rows * 2 * d) return;
  const int r = (int)(i / (2 * d));
  const int c = (int)(i - (long long)r * 2 * d);
  o[i] = (c < d) ? a[(long long)r * d + c] : b[(long long)r * d + (c - d)];
}

__global__ void concat3_k(const float* __restrict__ a, const float* __restrict__ b,
                          const float* __restrict__ c3, float* __restrict__ o,
                          int rows, int d)
{
  const long long i = (long long)blockIdx.x * blockDim.x + threadIdx.x;
  if (i >= (long long)rows * 3 * d) return;
  const int r = (int)(i / (3 * d));
  const int c = (int)(i - (long long)r * 3 * d);
  const float* src = (c < d) ? a : (c < 2 * d ? b : c3);
  const int cc = (c < d) ? c : (c < 2 * d ? c - d : c - 2 * d);
  o[i] = src[(long long)r * d + cc];
}

// ---------------------------------------------------------------------------
// Fused LayerNorm epilogues (D = 1024, block 256, 4 elems/thread).
// ---------------------------------------------------------------------------
__device__ __forceinline__ void ln_finish(float v[4], float sum, float sq,
                                          const float* __restrict__ g,
                                          const float* __restrict__ be,
                                          float* __restrict__ orow, int t)
{
  for (int o = 16; o; o >>= 1) { sum += __shfl_xor(sum, o, 32); sq += __shfl_xor(sq, o, 32); }
  __shared__ float s1[8], s2[8];
  const int lane = t & 31, wid = t >> 5;
  if (lane == 0) { s1[wid] = sum; s2[wid] = sq; }
  __syncthreads();
  sum = 0.f; sq = 0.f;
#pragma unroll
  for (int i = 0; i < 8; ++i) { sum += s1[i]; sq += s2[i]; }
  const float mean = sum * (1.0f / CD);
  const float var  = sq * (1.0f / CD) - mean * mean;
  const float rstd = rsqrtf(var + 1e-5f);
#pragma unroll
  for (int i = 0; i < 4; ++i) {
    const int c = t + i * 256;
    orow[c] = (v[i] - mean) * rstd * g[c] + be[c];
  }
}

__global__ __launch_bounds__(256) void ln_residual_k(
    const float* __restrict__ x, const float* __restrict__ y,
    const float* __restrict__ g, const float* __restrict__ be,
    float* __restrict__ o)
{
  const int r = blockIdx.x, t = threadIdx.x;
  float v[4]; float sum = 0.f, sq = 0.f;
#pragma unroll
  for (int i = 0; i < 4; ++i) {
    const int c = t + i * 256;
    const float a = x[(long long)r * CD + c] + y[(long long)r * CD + c];
    v[i] = a; sum += a; sq += a * a;
  }
  ln_finish(v, sum, sq, g, be, o + (long long)r * CD, t);
}

__global__ __launch_bounds__(256) void ln_gate_k(
    const float* __restrict__ gate, const float* __restrict__ a,
    const float* __restrict__ bm, const float* __restrict__ g,
    const float* __restrict__ be, float* __restrict__ o)
{
  const int r = blockIdx.x, t = threadIdx.x;
  float v[4]; float sum = 0.f, sq = 0.f;
#pragma unroll
  for (int i = 0; i < 4; ++i) {
    const int c = t + i * 256;
    const long long ix = (long long)r * CD + c;
    const float gg = gate[ix];
    const float val = gg * a[ix] + (1.f - gg) * bm[ix];
    v[i] = val; sum += val; sq += val * val;
  }
  ln_finish(v, sum, sq, g, be, o + (long long)r * CD, t);
}

// ---------------------------------------------------------------------------
// Host-side helpers
// ---------------------------------------------------------------------------
static inline void launch_gemm(hipStream_t st,
    const float* A, long long lda, long long sA,
    const float* W, long long ldw, long long sW,
    const float* bias, long long sBias,
    float* O, long long ldo, long long sO,
    int M, int N, int K, int batch, float alpha, int act, int nt)
{
  const int tmax = (N >= 64) ? 4 : 1;       // N is always 16, 64 or 1024
  dim3 grid(N / (16 * tmax), (M + 63) / 64, batch);
  dim3 blk(128);
#define GEMM_GO(T, NTF, ACT)                                                   \
  gemm_wmma_bf16_t<T, NTF, ACT><<<grid, blk, 0, st>>>(                         \
      A, lda, sA, W, ldw, sW, bias, sBias, O, ldo, sO, M, N, K, alpha)
  if (nt) {
    if (tmax == 4) {
      if (act == 0)      GEMM_GO(4, true, 0);
      else if (act == 1) GEMM_GO(4, true, 1);
      else               GEMM_GO(4, true, 2);
    } else {
      GEMM_GO(1, true, 0);                  // scores: N = 16
    }
  } else {
    GEMM_GO(4, false, 0);                   // NN GEMMs always have N = 1024
  }
#undef GEMM_GO
}

// Single-query MHA with K/V projections factored out (see header comment).
static void attention_block(hipStream_t st,
    const float* query,                 // [B, D]
    const float* X, int Sk,             // [B, Sk, D]
    const int* mask,                    // [B, Sk] or null
    const float* qkv_w, const float* qkv_b,
    const float* out_w, const float* out_b,
    float* qh, float* T, float* sc, float* attn, float* U, float* ho,
    float* outbuf, float* mean_out)
{
  const long long D = CD, NH = CNH;
  // qh = query @ Wq.T + bq
  launch_gemm(st, query, D, 0, qkv_w, D, 0, qkv_b, 0,
              qh, D, 0, CB, CD, CD, 1, 1.f, 0, 1);
  // t[b,h,:] = qh[b,h,:] @ Wk_h  (per-head NN, batched over heads)
  launch_gemm(st, qh, D, 64, qkv_w + D * D, D, 64 * D, nullptr, 0,
              T, NH * D, D, CB, CD, 64, CNH, 1.f, 0, 0);
  // scores[b,j,h] = (X_b[j,:] . T_b[h,:]) / sqrt(64)   (batched over b, N=16)
  launch_gemm(st, X, D, (long long)Sk * D, T, D, NH * D, nullptr, 0,
              sc, CNH, (long long)Sk * CNH, Sk, CNH, CD, CB, 0.125f, 0, 1);
  softmax16_k<<<dim3(CNH / 4, CB), dim3(128), 0, st>>>(sc, mask, attn, Sk);
  if (mean_out)
    attn_mean16_k<<<(CB * Sk + 255) / 256, 256, 0, st>>>(attn, mean_out, Sk);
  // u[b,h,:] = attn[b,h,:] @ X_b   (batched over b, M=16, NN)
  launch_gemm(st, attn, Sk, (long long)CNH * Sk, X, D, (long long)Sk * D,
              nullptr, 0, U, D, NH * D, CNH, CD, Sk, CB, 1.f, 0, 0);
  // ho[b, 64h+n] = Wv[64h+n,:] . u[b,h,:] + bv[64h+n]  (batched over heads)
  launch_gemm(st, U, NH * D, D, qkv_w + 2 * D * D, D, 64 * D,
              qkv_b + 2 * CD, 64, ho, D, 64, CB, 64, CD, CNH, 1.f, 0, 1);
  // out = ho @ Wout.T + bout
  launch_gemm(st, ho, D, 0, out_w, D, 0, out_b, 0,
              outbuf, D, 0, CB, CD, CD, 1, 1.f, 0, 1);
}

extern "C" void kernel_launch(void* const* d_in, const int* in_sizes, int n_in,
                              void* d_out, int out_size, void* d_ws, size_t ws_size,
                              hipStream_t stream)
{
  (void)in_sizes; (void)n_in; (void)out_size; (void)ws_size;
  const float* control = (const float*)d_in[0];
  const float* memory  = (const float*)d_in[1];
  const float* qe      = (const float*)d_in[2];
  const float* objf    = (const float*)d_in[3];
  const float* relf    = (const float*)d_in[4];
  const int*   qmask   = (const int*)d_in[5];

  const float* ctrl_qkv_w  = (const float*)d_in[6];
  const float* ctrl_qkv_b  = (const float*)d_in[7];
  const float* ctrl_out_w  = (const float*)d_in[8];
  const float* ctrl_out_b  = (const float*)d_in[9];
  const float* ctrl_up_w1  = (const float*)d_in[10];
  const float* ctrl_up_b1  = (const float*)d_in[11];
  const float* ctrl_up_w2  = (const float*)d_in[12];
  const float* ctrl_up_b2  = (const float*)d_in[13];
  const float* ctrl_proj_w = (const float*)d_in[14];
  const float* ctrl_proj_b = (const float*)d_in[15];
  const float* ctrl_ln_g   = (const float*)d_in[16];
  const float* ctrl_ln_b   = (const float*)d_in[17];
  const float* obj_qkv_w   = (const float*)d_in[18];
  const float* obj_qkv_b   = (const float*)d_in[19];
  const float* obj_out_w   = (const float*)d_in[20];
  const float* obj_out_b   = (const float*)d_in[21];
  const float* rel_qkv_w   = (const float*)d_in[22];
  const float* rel_qkv_b   = (const float*)d_in[23];
  const float* rel_out_w   = (const float*)d_in[24];
  const float* rel_out_b   = (const float*)d_in[25];
  const float* rd_comb_w1  = (const float*)d_in[26];
  const float* rd_comb_b1  = (const float*)d_in[27];
  const float* rd_comb_w2  = (const float*)d_in[28];
  const float* rd_comb_b2  = (const float*)d_in[29];
  const float* rd_gate_w   = (const float*)d_in[30];
  const float* rd_gate_b   = (const float*)d_in[31];
  const float* rd_ln_g     = (const float*)d_in[32];
  const float* rd_ln_b     = (const float*)d_in[33];
  const float* wr_gate_w   = (const float*)d_in[34];
  const float* wr_gate_b   = (const float*)d_in[35];
  const float* wr_tr_w1    = (const float*)d_in[36];
  const float* wr_tr_b1    = (const float*)d_in[37];
  const float* wr_tr_w2    = (const float*)d_in[38];
  const float* wr_tr_b2    = (const float*)d_in[39];
  const float* wr_ln_g     = (const float*)d_in[40];
  const float* wr_ln_b     = (const float*)d_in[41];

  // Outputs (tuple order): new_control, new_memory, obj_attn, rel_attn
  float* newc  = (float*)d_out;
  float* newm  = newc + (long long)CB * CD;
  float* oattn = newm + (long long)CB * CD;
  float* rattn = oattn + (long long)CB * CNO;

  // Workspace bump allocator (~68 MB of f32)
  float* wsp = (float*)d_ws;
  size_t off = 0;
  auto alloc = [&](size_t n) { float* p = wsp + off; off += (n + 63) & ~(size_t)63; return p; };
  float* qh       = alloc((size_t)CB * CD);
  float* T        = alloc((size_t)CB * CNH * CD);
  float* sc       = alloc((size_t)CB * CNR * CNH);   // sized for largest (rel)
  float* attn     = alloc((size_t)CB * CNH * CNR);
  float* U        = alloc((size_t)CB * CNH * CD);
  float* ho       = alloc((size_t)CB * CD);
  float* attq     = alloc((size_t)CB * CD);
  float* cat2     = alloc((size_t)CB * 2 * CD);
  float* cat3     = alloc((size_t)CB * 3 * CD);
  float* h1       = alloc((size_t)CB * CD);
  float* t2       = alloc((size_t)CB * CD);
  float* proj     = alloc((size_t)CB * CD);
  float* obj_info = alloc((size_t)CB * CD);
  float* rel_info = alloc((size_t)CB * CD);
  float* combined = alloc((size_t)CB * CD);
  float* gate     = alloc((size_t)CB * CD);
  float* read_out = alloc((size_t)CB * CD);
  float* wgv      = alloc((size_t)CB * CD);
  float* update   = alloc((size_t)CB * CD);

  const int THR = 256;
  const long long n2 = (long long)CB * 2 * CD;
  const long long n3 = (long long)CB * 3 * CD;

  // ---------------- Control unit ----------------
  attention_block(stream, control, qe, CS, qmask,
                  ctrl_qkv_w, ctrl_qkv_b, ctrl_out_w, ctrl_out_b,
                  qh, T, sc, attn, U, ho, attq, nullptr);
  concat2_k<<<(int)((n2 + THR - 1) / THR), THR, 0, stream>>>(control, attq, cat2, CB, CD);
  launch_gemm(stream, cat2, 2 * CD, 0, ctrl_up_w1, 2 * CD, 0, ctrl_up_b1, 0,
              h1, CD, 0, CB, CD, 2 * CD, 1, 1.f, 1, 1);                   // GELU
  launch_gemm(stream, h1, CD, 0, ctrl_up_w2, CD, 0, ctrl_up_b2, 0,
              t2, CD, 0, CB, CD, CD, 1, 1.f, 0, 1);
  launch_gemm(stream, attq, CD, 0, ctrl_proj_w, CD, 0, ctrl_proj_b, 0,
              proj, CD, 0, CB, CD, CD, 1, 1.f, 0, 1);
  ln_residual_k<<<CB, 256, 0, stream>>>(t2, proj, ctrl_ln_g, ctrl_ln_b, newc);

  // ---------------- Read unit: object & relation attention ----------------
  attention_block(stream, newc, objf, CNO, nullptr,
                  obj_qkv_w, obj_qkv_b, obj_out_w, obj_out_b,
                  qh, T, sc, attn, U, ho, obj_info, oattn);
  attention_block(stream, newc, relf, CNR, nullptr,
                  rel_qkv_w, rel_qkv_b, rel_out_w, rel_out_b,
                  qh, T, sc, attn, U, ho, rel_info, rattn);

  concat2_k<<<(int)((n2 + THR - 1) / THR), THR, 0, stream>>>(obj_info, rel_info, cat2, CB, CD);
  launch_gemm(stream, cat2, 2 * CD, 0, rd_comb_w1, 2 * CD, 0, rd_comb_b1, 0,
              h1, CD, 0, CB, CD, 2 * CD, 1, 1.f, 1, 1);                   // GELU
  launch_gemm(stream, h1, CD, 0, rd_comb_w2, CD, 0, rd_comb_b2, 0,
              combined, CD, 0, CB, CD, CD, 1, 1.f, 0, 1);
  concat2_k<<<(int)((n2 + THR - 1) / THR), THR, 0, stream>>>(combined, memory, cat2, CB, CD);
  launch_gemm(stream, cat2, 2 * CD, 0, rd_gate_w, 2 * CD, 0, rd_gate_b, 0,
              gate, CD, 0, CB, CD, 2 * CD, 1, 1.f, 2, 1);                 // sigmoid
  ln_gate_k<<<CB, 256, 0, stream>>>(gate, combined, memory, rd_ln_g, rd_ln_b, read_out);

  // ---------------- Write unit ----------------
  concat3_k<<<(int)((n3 + THR - 1) / THR), THR, 0, stream>>>(memory, read_out, newc, cat3, CB, CD);
  launch_gemm(stream, cat3, 3 * CD, 0, wr_gate_w, 3 * CD, 0, wr_gate_b, 0,
              wgv, CD, 0, CB, CD, 3 * CD, 1, 1.f, 2, 1);                  // sigmoid
  concat2_k<<<(int)((n2 + THR - 1) / THR), THR, 0, stream>>>(read_out, newc, cat2, CB, CD);
  launch_gemm(stream, cat2, 2 * CD, 0, wr_tr_w1, 2 * CD, 0, wr_tr_b1, 0,
              h1, CD, 0, CB, CD, 2 * CD, 1, 1.f, 1, 1);                   // GELU
  launch_gemm(stream, h1, CD, 0, wr_tr_w2, CD, 0, wr_tr_b2, 0,
              update, CD, 0, CB, CD, CD, 1, 1.f, 0, 1);
  ln_gate_k<<<CB, 256, 0, stream>>>(wgv, update, memory, wr_ln_g, wr_ln_b, newm);
}
